// CharRNN_86792699118068
// MI455X (gfx1250) — compile-verified, matrix operand order checked by reading, not run on hardware
//
#include <hip/hip_runtime.h>

// ---------------- problem constants ----------------
constexpr int VOCAB  = 96;
constexpr int EMBED  = 128;
constexpr int HIDDEN = 2048;
constexpr int BATCH  = 128;
constexpr int SEQ    = 512;

// ---------------- vector types ----------------
typedef __attribute__((ext_vector_type(16))) __bf16        v16bf;
typedef __attribute__((ext_vector_type(8)))  __bf16        v8bf;
typedef __attribute__((ext_vector_type(8)))  float         v8f;
typedef __attribute__((ext_vector_type(4)))  unsigned int  v4u;
typedef __attribute__((ext_vector_type(8)))  int           v8i;
typedef __attribute__((ext_vector_type(4)))  int           v4i;

// Load a bf16 WMMA operand fragment slice for this lane.
// Per CDNA5 ISA 16-bit A/B layout (wave32): a lane holds 8 contiguous K values
// starting at kOff (0 for lanes 0-15, 8 for lanes 16-31) and another 8 at +16.
// p must already include the row base and kOff. Works for global or LDS
// pointers (address-space inference picks ds_load vs global_load).
__device__ inline v16bf load_frag(const __bf16* p) {
    v8bf lo = *(const v8bf*)(p);        // 16B load
    v8bf hi = *(const v8bf*)(p + 16);   // 16B load
    return __builtin_shufflevector(lo, hi,
        0, 1, 2, 3, 4, 5, 6, 7, 8, 9, 10, 11, 12, 13, 14, 15);
}

__device__ inline v8f wmma_bf16(v16bf a, v16bf b, v8f c) {
    return __builtin_amdgcn_wmma_f32_16x16x32_bf16(
        /*neg_a=*/false, a, /*neg_b=*/false, b,
        /*c_mod=*/(short)0, c, /*reuse_a=*/false, /*reuse_b=*/false);
}

// ---------------- prep: f32 -> bf16 ----------------
__global__ void f32_to_bf16_kernel(const float* __restrict__ src,
                                   __bf16* __restrict__ dst, int n) {
    int i = blockIdx.x * blockDim.x + threadIdx.x;
    if (i < n) dst[i] = (__bf16)src[i];
}

// ---------------- shared logits tile (16x16, K = HIDDEN) ----------------
// C[m][v] = sum_k h[m][k] * W_hy[v][k] + b_y[v]   for one 16x16 tile
__device__ inline void logits_tile(int lw, int lane, int tlog,
                                   const __bf16* __restrict__ h,
                                   const __bf16* __restrict__ why,
                                   const float* __restrict__ by,
                                   float* __restrict__ out) {
    const int mt = lw / (VOCAB / 16);       // 0..7
    const int vt = lw % (VOCAB / 16);       // 0..5
    const int mbase = mt * 16;
    const int nbase = vt * 16;
    const int row   = lane & 15;
    const int kOff  = (lane >> 4) * 8;

    const __bf16* Ap = h   + (size_t)(mbase + row) * HIDDEN + kOff;
    const __bf16* Bp = why + (size_t)(nbase + row) * HIDDEN + kOff;

    v8f c = {};
    for (int k = 0; k < HIDDEN; k += 32) {
        v16bf a = load_frag(Ap + k);
        v16bf b = load_frag(Bp + k);
        c = wmma_bf16(a, b, c);
    }

    const int   n    = nbase + row;
    const float bias = by[n];
    const int   mAdd = (lane >> 4) * 8;
#pragma unroll
    for (int r = 0; r < 8; ++r) {
        int m = mbase + mAdd + r;
        out[(size_t)(m * SEQ + tlog) * VOCAB + n] = c[r] + bias;
    }
}

// ---------------- per-timestep kernel ----------------
// Blocks 0..31  (waves 0..255)  : h_t = tanh(b_h + embed(x_t)@W_xh^T + h_{t-1}@W_hh^T)
//   - wave 0 of each block DMAs the 16-row A panel (64 KB of h_{t-1}) into LDS
//     via the Tensor Data Mover; all 8 waves share the M-tile, reading A via
//     ds_load after the barrier.
//   - each wave computes a 16(M) x 64(N) tile (4 f32 accumulators).
// Blocks 32..37 (waves 256..303): logits_{t-1} = h_{t-1} @ W_hy^T + b_y (t>0)
constexpr int H_WAVES     = (BATCH / 16) * (HIDDEN / 64);      // 8 * 32 = 256
constexpr int H_BLOCKS    = H_WAVES / 8;                       // 32
constexpr int LOG_WAVES   = (BATCH / 16) * (VOCAB / 16);       // 8 * 6  = 48
constexpr int STEP_BLOCKS = H_BLOCKS + LOG_WAVES / 8;          // 38

__global__ void __launch_bounds__(256)
rnn_step_kernel(int t,
                const __bf16* __restrict__ hPrev,
                __bf16*       __restrict__ hCur,
                const __bf16* __restrict__ whh,
                const __bf16* __restrict__ wxh,
                const __bf16* __restrict__ emb,
                const __bf16* __restrict__ why,
                const float*  __restrict__ bh,
                const float*  __restrict__ by,
                const int*    __restrict__ x,
                float*        __restrict__ out) {
    __shared__ __bf16 smem[16 * HIDDEN];   // 64 KB A panel (CDNA5: 320 KB/WGP)

    const int lane = threadIdx.x & 31;

    if (blockIdx.x < H_BLOCKS) {
        const int mt    = blockIdx.x >> 2;         // 4 blocks per M-tile
        const int mbase = mt * 16;

        // ---- TDM: async-copy the contiguous 16 x 2048 bf16 A panel to LDS ----
        if ((threadIdx.x >> 5) == 0) {             // wave 0 issues the DMA
            constexpr unsigned ELEMS = 16u * HIDDEN * 2u / 8u;   // 8192 x 8B = 64KB
            const unsigned long long ga =
                (unsigned long long)(uintptr_t)(hPrev + (size_t)mbase * HIDDEN);
            // Flat LDS address truncates to the LDS byte offset (ISA 10.2).
            const unsigned lds = (unsigned)(uintptr_t)(void*)&smem[0];

            // D# group 0: count=1 | lds_addr | global_addr[56:0] | type=2
            v4u g0 = {};
            g0[0] = 1u;
            g0[1] = lds;
            g0[2] = (unsigned)ga;
            g0[3] = (unsigned)((ga >> 32) & 0x01FFFFFFu) | (2u << 30);

            // D# group 1: wg_mask=0, data_size=3 (8B), 1-D tensor of 8192 elems
            v8i g1 = {};
            g1[0] = (int)(3u << 16);                         // data_size = 8B
            g1[1] = (int)((ELEMS & 0xFFFFu) << 16);          // tensor_dim0[15:0]
            g1[2] = (int)((ELEMS >> 16) | (1u << 16));       // dim0[31:16], tensor_dim1=1
            g1[3] = (int)((ELEMS & 0xFFFFu) << 16);          // tile_dim0 = 8192
            g1[4] = 0;                                       // tile_dim1/2 unused
            g1[5] = (int)ELEMS;                              // tensor_dim0_stride lo
            g1[6] = 0;
            g1[7] = 0;

            v4i gz  = {};                                    // groups 2/3: dims >=2 unused
            v8i gz8 = {};                                    // trailing group (unused)
            __builtin_amdgcn_tensor_load_to_lds(g0, g1, gz, gz, gz8, 0);
            __builtin_amdgcn_s_wait_tensorcnt(0);            // drain TENSORcnt
        }
        __syncthreads();                                     // release consumer waves

        const int wid   = blockIdx.x * 8 + (threadIdx.x >> 5);
        const int ng    = wid & 31;                // 0..31
        const int nbase = ng * 64;
        const int row   = lane & 15;
        const int kOff  = (lane >> 4) * 8;

        const __bf16* As  = &smem[row * HIDDEN + kOff];             // LDS
        const __bf16* Bp0 = whh + (size_t)(nbase +  0 + row) * HIDDEN + kOff;
        const __bf16* Bp1 = whh + (size_t)(nbase + 16 + row) * HIDDEN + kOff;
        const __bf16* Bp2 = whh + (size_t)(nbase + 32 + row) * HIDDEN + kOff;
        const __bf16* Bp3 = whh + (size_t)(nbase + 48 + row) * HIDDEN + kOff;

        v8f c0 = {}, c1 = {}, c2 = {}, c3 = {};

        // ---- recurrent term: h_{t-1} @ W_hh^T (K = 2048) ----
        for (int k = 0; k < HIDDEN; k += 32) {
            v16bf a = load_frag(As + k);           // ds_load_b128 x2
            c0 = wmma_bf16(a, load_frag(Bp0 + k), c0);
            c1 = wmma_bf16(a, load_frag(Bp1 + k), c1);
            c2 = wmma_bf16(a, load_frag(Bp2 + k), c2);
            c3 = wmma_bf16(a, load_frag(Bp3 + k), c3);
        }

        // ---- input term: embed(x[m, t]) @ W_xh^T (K = 128) ----
        const int id = x[(mbase + row) * SEQ + t];
        const __bf16* Xp = emb + (size_t)id * EMBED + kOff;
        const __bf16* W0 = wxh + (size_t)(nbase +  0 + row) * EMBED + kOff;
        const __bf16* W1 = wxh + (size_t)(nbase + 16 + row) * EMBED + kOff;
        const __bf16* W2 = wxh + (size_t)(nbase + 32 + row) * EMBED + kOff;
        const __bf16* W3 = wxh + (size_t)(nbase + 48 + row) * EMBED + kOff;
#pragma unroll
        for (int k = 0; k < EMBED; k += 32) {
            v16bf a = load_frag(Xp + k);
            c0 = wmma_bf16(a, load_frag(W0 + k), c0);
            c1 = wmma_bf16(a, load_frag(W1 + k), c1);
            c2 = wmma_bf16(a, load_frag(W2 + k), c2);
            c3 = wmma_bf16(a, load_frag(W3 + k), c3);
        }

        // ---- epilogue: bias + tanh, write bf16 h_t ----
        const int mAdd = (lane >> 4) * 8;
#pragma unroll
        for (int j = 0; j < 4; ++j) {
            v8f cj = (j == 0) ? c0 : (j == 1) ? c1 : (j == 2) ? c2 : c3;
            const int n = nbase + j * 16 + row;
            const float bias = bh[n];
#pragma unroll
            for (int r = 0; r < 8; ++r) {
                int m = mbase + mAdd + r;
                float v = tanhf(cj[r] + bias);
                hCur[(size_t)m * HIDDEN + n] = (__bf16)v;
            }
        }
    } else if (t > 0) {
        const int wid = blockIdx.x * 8 + (threadIdx.x >> 5);
        logits_tile(wid - H_WAVES, lane, t - 1, hPrev, why, by, out);
    }
}

// ---------------- final kernel: logits for t = SEQ-1 + final hidden f32 ----
constexpr int COPY_WAVES   = (BATCH * HIDDEN) / 256;           // 1024 waves
constexpr int FINAL_WAVES  = LOG_WAVES + COPY_WAVES;           // 1072
constexpr int FINAL_BLOCKS = FINAL_WAVES / 8;                  // 134

__global__ void __launch_bounds__(256)
rnn_final_kernel(const __bf16* __restrict__ hFinal,
                 const __bf16* __restrict__ why,
                 const float*  __restrict__ by,
                 float*        __restrict__ out) {
    const int wid  = blockIdx.x * (blockDim.x >> 5) + (threadIdx.x >> 5);
    const int lane = threadIdx.x & 31;

    if (wid < LOG_WAVES) {
        logits_tile(wid, lane, SEQ - 1, hFinal, why, by, out);
    } else {
        // copy final hidden (bf16) -> f32 tail of d_out
        const int cw   = wid - LOG_WAVES;          // 0..1023
        const int base = cw * 256;
        float* hOut = out + (size_t)BATCH * SEQ * VOCAB;
#pragma unroll
        for (int i = 0; i < 8; ++i) {
            int idx = base + i * 32 + lane;
            hOut[idx] = (float)hFinal[idx];
        }
    }
}

// ---------------- workspace layout (bytes) ----------------
constexpr size_t WS_WHH  = 0;
constexpr size_t WS_WXH  = WS_WHH + (size_t)HIDDEN * HIDDEN * 2;   //  8388608
constexpr size_t WS_WHY  = WS_WXH + (size_t)HIDDEN * EMBED  * 2;   // +524288
constexpr size_t WS_EMB  = WS_WHY + (size_t)VOCAB  * HIDDEN * 2;   // +393216
constexpr size_t WS_H0   = WS_EMB + (size_t)VOCAB  * EMBED  * 2;   // +24576
constexpr size_t WS_H1   = WS_H0  + (size_t)BATCH  * HIDDEN * 2;   // +524288
// total ~10.4 MB

extern "C" void kernel_launch(void* const* d_in, const int* in_sizes, int n_in,
                              void* d_out, int out_size, void* d_ws, size_t ws_size,
                              hipStream_t stream) {
    const int*   x      = (const int*)  d_in[0];
    const float* hidden = (const float*)d_in[1];
    const float* emb    = (const float*)d_in[2];
    const float* wxh    = (const float*)d_in[3];
    const float* whh    = (const float*)d_in[4];
    const float* bh     = (const float*)d_in[5];
    const float* why    = (const float*)d_in[6];
    const float* by     = (const float*)d_in[7];
    float* out = (float*)d_out;

    char* ws = (char*)d_ws;
    __bf16* whh_b = (__bf16*)(ws + WS_WHH);
    __bf16* wxh_b = (__bf16*)(ws + WS_WXH);
    __bf16* why_b = (__bf16*)(ws + WS_WHY);
    __bf16* emb_b = (__bf16*)(ws + WS_EMB);
    __bf16* h0    = (__bf16*)(ws + WS_H0);
    __bf16* h1    = (__bf16*)(ws + WS_H1);

    auto conv = [&](const float* s, __bf16* d, int n) {
        f32_to_bf16_kernel<<<(n + 255) / 256, 256, 0, stream>>>(s, d, n);
    };
    conv(whh,    whh_b, HIDDEN * HIDDEN);
    conv(wxh,    wxh_b, HIDDEN * EMBED);
    conv(why,    why_b, VOCAB  * HIDDEN);
    conv(emb,    emb_b, VOCAB  * EMBED);
    conv(hidden, h0,    BATCH  * HIDDEN);

    __bf16* bufs[2] = { h0, h1 };
    for (int t = 0; t < SEQ; ++t) {
        rnn_step_kernel<<<STEP_BLOCKS, 256, 0, stream>>>(
            t, bufs[t & 1], bufs[(t + 1) & 1],
            whh_b, wxh_b, emb_b, why_b, bh, by, x, out);
    }
    // h_{511} lands in bufs[(511+1)&1] == bufs[0]
    rnn_final_kernel<<<FINAL_BLOCKS, 256, 0, stream>>>(bufs[0], why_b, by, out);
}
